// AttnOutputDecoder_61125974556735
// MI455X (gfx1250) — compile-verified, matrix operand order checked by reading, not run on hardware
//
#include <hip/hip_runtime.h>
#include <hip/hip_bf16.h>
#include <math.h>

// ---- sizes ----
#define BB 16
#define TT 64
#define SS 128
#define DD 512
#define VV 32000
#define NWG 16   // persistent LSTM workgroups

typedef __attribute__((ext_vector_type(16))) __bf16 v16bf;
typedef __attribute__((ext_vector_type(8)))  float  v8f;
typedef __attribute__((ext_vector_type(4)))  unsigned int u32x4;

struct FragBits { u32x4 a, b; };   // 32B, trivially copyable

static __device__ __forceinline__ v16bf load_frag(const __bf16* p) {
  FragBits t;
  t.a = *(const u32x4*)p;          // K run [half*8 .. half*8+7]
  t.b = *(const u32x4*)(p + 16);   // K run [16+half*8 .. 16+half*8+7]
  return __builtin_bit_cast(v16bf, t);
}

#define WMMA_BF16(a,b,c) \
  __builtin_amdgcn_wmma_f32_16x16x32_bf16(false,(a),false,(b),(short)0,(c),false,false)

#define V8F_ZERO {0.f,0.f,0.f,0.f,0.f,0.f,0.f,0.f}

__device__ __forceinline__ float sigf(float x) { return 1.0f / (1.0f + __expf(-x)); }

// ---------------------------------------------------------------------------
// Generic bf16 WMMA GEMM: C[M,N] = A[M,K] * B[N,K]^T (+ bias[n])
// Requires M%128==0, N%128==0, K%32==0 (true for all call sites here).
// 256 threads = 8 waves; block tile 128x128; K-step 32; LDS-staged.
// Accumulators are 8 named v8f registers (no arrays -> no scratch spill).
// ---------------------------------------------------------------------------
__global__ __launch_bounds__(256) void k_gemm_bf16(
    const __bf16* __restrict__ A, const __bf16* __restrict__ Bm,
    const float* __restrict__ bias, float* __restrict__ Cf,
    __bf16* __restrict__ Cb, int M, int N, int K)
{
  __shared__ __attribute__((aligned(16))) __bf16 As[128 * 40]; // 32 data + 8 pad
  __shared__ __attribute__((aligned(16))) __bf16 Bs[128 * 40];

  const int tid  = threadIdx.x;
  const int lane = tid & 31;
  const int wv   = tid >> 5;
  const int half = lane >> 4;
  const int l16  = lane & 15;
  const int m0   = blockIdx.y * 128;
  const int n0   = blockIdx.x * 128;
  const int mw   = (wv >> 1) * 32;  // wave m offset (4 rows of waves)
  const int nw   = (wv & 1) * 64;   // wave n offset (2 cols of waves)

  v8f c00 = V8F_ZERO, c01 = V8F_ZERO, c02 = V8F_ZERO, c03 = V8F_ZERO;
  v8f c10 = V8F_ZERO, c11 = V8F_ZERO, c12 = V8F_ZERO, c13 = V8F_ZERO;

  for (int k0 = 0; k0 < K; k0 += 32) {
    // cooperative stage: threads 0-127 copy one A row (64B), 128-255 one B row
    {
      const __bf16* src = (tid < 128)
          ? (A  + (size_t)(m0 + tid)         * K + k0)
          : (Bm + (size_t)(n0 + (tid - 128)) * K + k0);
      __bf16* dst = (tid < 128) ? (As + tid * 40) : (Bs + (tid - 128) * 40);
      const u32x4* s4 = (const u32x4*)src;
      u32x4* d4 = (u32x4*)dst;
      u32x4 t0 = s4[0], t1 = s4[1], t2 = s4[2], t3 = s4[3];
      d4[0] = t0; d4[1] = t1; d4[2] = t2; d4[3] = t3;
    }
    __syncthreads();

    const __bf16* pA = As + (mw + l16) * 40 + half * 8;
    const __bf16* pB = Bs + (nw + l16) * 40 + half * 8;
    v16bf a0 = load_frag(pA);
    v16bf a1 = load_frag(pA + 16 * 40);
    v16bf b0 = load_frag(pB);
    v16bf b1 = load_frag(pB + 16 * 40);
    v16bf b2 = load_frag(pB + 32 * 40);
    v16bf b3 = load_frag(pB + 48 * 40);

    c00 = WMMA_BF16(a0, b0, c00);
    c01 = WMMA_BF16(a0, b1, c01);
    c02 = WMMA_BF16(a0, b2, c02);
    c03 = WMMA_BF16(a0, b3, c03);
    c10 = WMMA_BF16(a1, b0, c10);
    c11 = WMMA_BF16(a1, b1, c11);
    c12 = WMMA_BF16(a1, b2, c12);
    c13 = WMMA_BF16(a1, b3, c13);
    __syncthreads();
  }

#define EPILOGUE(ACC, I, J)                                      \
  {                                                              \
    const int n = n0 + nw + (J) * 16 + l16;                      \
    const float bv = bias ? bias[n] : 0.0f;                      \
    _Pragma("unroll")                                            \
    for (int r = 0; r < 8; ++r) {                                \
      const int m = m0 + mw + (I) * 16 + half * 8 + r;           \
      const float val = (ACC)[r] + bv;                           \
      if (Cf) Cf[(size_t)m * N + n] = val;                       \
      if (Cb) Cb[(size_t)m * N + n] = (__bf16)val;               \
    }                                                            \
  }
  EPILOGUE(c00, 0, 0) EPILOGUE(c01, 0, 1) EPILOGUE(c02, 0, 2) EPILOGUE(c03, 0, 3)
  EPILOGUE(c10, 1, 0) EPILOGUE(c11, 1, 1) EPILOGUE(c12, 1, 2) EPILOGUE(c13, 1, 3)
#undef EPILOGUE
}

// ---------------------------------------------------------------------------
// Persistent LSTM: 16 WGs, each owns d-slice [wg*32, wg*32+32) and computes
// all 4 gate columns for it (gemm 16x128 vs h[16x512]); cell update WG-local
// via LDS; one grid barrier per step; h double-buffered.
// ---------------------------------------------------------------------------
__global__ __launch_bounds__(256) void k_lstm(
    const float* __restrict__ gates_x, const __bf16* __restrict__ Whh,
    __bf16* __restrict__ h2, float* __restrict__ c,
    __bf16* __restrict__ hs, unsigned* __restrict__ bar)
{
  const int wg   = blockIdx.x;
  const int tid  = threadIdx.x;
  const int lane = tid & 31;
  const int wv   = tid >> 5;
  const int half = lane >> 4;
  const int l16  = lane & 15;
  const int g    = wv >> 1;              // gate index 0..3 (i,f,g,o)
  const int sub  = wv & 1;               // which 16-wide half of the 32 d's
  const int ncol0 = g * 512 + wg * 32 + sub * 16;

  __shared__ float gl[8 * 256];          // 8 waves x 16x16 f32 tiles

  unsigned cnt = 0;
  for (int t = 0; t < TT; ++t) {
    const __bf16* hin = h2 + (size_t)(t & 1) * (BB * DD);

    v8f acc = V8F_ZERO;
#pragma unroll 4
    for (int kt = 0; kt < 16; ++kt) {
      const int k0 = kt * 32 + half * 8;
      v16bf af = load_frag(hin + (size_t)l16 * DD + k0);             // A: h
      v16bf bf = load_frag(Whh + (size_t)(ncol0 + l16) * DD + k0);   // B: W_hh
      acc = WMMA_BF16(af, bf, acc);
    }

    // exchange gate tiles through LDS: element (m = half*8+r, n = l16)
#pragma unroll
    for (int r = 0; r < 8; ++r)
      gl[wv * 256 + (half * 8 + r) * 16 + l16] = acc[r];
    __syncthreads();

    // cell update for this WG's d-slice (16 b x 32 d = 512 elems)
    for (int e = tid; e < 512; e += 256) {
      const int b  = e >> 5;
      const int dd = e & 31;
      const int d  = wg * 32 + dd;
      const int tv = dd >> 4;
      const int nn = dd & 15;
      const float* gx = gates_x + ((size_t)b * TT + t) * 2048;
      const float iv = gl[(0 + tv) * 256 + b * 16 + nn] + gx[d];
      const float fv = gl[(2 + tv) * 256 + b * 16 + nn] + gx[512 + d];
      const float gv = gl[(4 + tv) * 256 + b * 16 + nn] + gx[1024 + d];
      const float ov = gl[(6 + tv) * 256 + b * 16 + nn] + gx[1536 + d];
      const float cv = c[b * DD + d];
      const float cn = sigf(fv) * cv + sigf(iv) * tanhf(gv);
      const float hv = sigf(ov) * tanhf(cn);
      c[b * DD + d] = cn;
      const __bf16 hb = (__bf16)hv;
      h2[(size_t)((t + 1) & 1) * (BB * DD) + b * DD + d] = hb;
      hs[((size_t)b * TT + t) * DD + d] = hb;
    }
    __syncthreads();

    // grid barrier (monotonic counter, zeroed by init kernel each call)
    ++cnt;
    if (tid == 0) {
      __hip_atomic_fetch_add(bar, 1u, __ATOMIC_ACQ_REL, __HIP_MEMORY_SCOPE_AGENT);
      while (__hip_atomic_load(bar, __ATOMIC_ACQUIRE, __HIP_MEMORY_SCOPE_AGENT) <
             cnt * (unsigned)NWG)
        __builtin_amdgcn_s_sleep(8);
    }
    __syncthreads();
  }
}

// ---------------------------------------------------------------------------
// Scores + softmax fused: block = (b,t); e[s] = sum_d vt[d]*tanh(wh+ws),
// softmax over s=0..127, write a[bt*128 + s].
// ---------------------------------------------------------------------------
__global__ __launch_bounds__(256) void k_attn(
    const float* __restrict__ wh, const float* __restrict__ wsb,
    const float* __restrict__ vt, float* __restrict__ a)
{
  const int bt   = blockIdx.x;
  const int b    = bt >> 6;
  const int tid  = threadIdx.x;
  const int lane = tid & 31;
  const int wv   = tid >> 5;

  float wsr[16], vtr[16];
  const float* wsp = wsb + (size_t)bt * DD;
#pragma unroll
  for (int j = 0; j < 16; ++j) {
    const int d = lane + 32 * j;
    wsr[j] = wsp[d];
    vtr[j] = vt[d];
  }

  __shared__ float e[SS];
  for (int i = 0; i < 16; ++i) {
    const int s = wv + 8 * i;
    const float* whp = wh + ((size_t)b * SS + s) * DD;
    float sum = 0.0f;
#pragma unroll
    for (int j = 0; j < 16; ++j) {
      const int d = lane + 32 * j;
      sum += vtr[j] * tanhf(whp[d] + wsr[j]);
    }
    for (int off = 16; off; off >>= 1) sum += __shfl_xor(sum, off, 32);
    if (lane == 0) e[s] = sum;
  }
  __syncthreads();

  __shared__ float smax, ssum;
  if (wv == 0) {
    float m = -1e30f;
#pragma unroll
    for (int k = 0; k < 4; ++k) m = fmaxf(m, e[lane * 4 + k]);
    for (int off = 16; off; off >>= 1) m = fmaxf(m, __shfl_xor(m, off, 32));
    float s = 0.0f;
#pragma unroll
    for (int k = 0; k < 4; ++k) s += __expf(e[lane * 4 + k] - m);
    for (int off = 16; off; off >>= 1) s += __shfl_xor(s, off, 32);
    if (lane == 0) { smax = m; ssum = s; }
  }
  __syncthreads();
  if (tid < SS) a[(size_t)bt * SS + tid] = __expf(e[tid] - smax) / ssum;
}

// context[bt,d] = sum_s a[bt,s]*enc[b,s,d];  cat = [bf16(ctx) | hs_bf16]
__global__ void k_ctx(const float* __restrict__ a, const float* __restrict__ enc,
                      const __bf16* __restrict__ hs, __bf16* __restrict__ cat)
{
  const int idx = blockIdx.x * 256 + threadIdx.x;   // 1024*512
  const int bt = idx >> 9, d = idx & 511, b = bt >> 6;
  const float* ap = a + (size_t)bt * SS;
  const float* ep = enc + (size_t)b * SS * DD + d;
  float ctx = 0.0f;
  for (int s = 0; s < SS; ++s) ctx += ap[s] * ep[(size_t)s * DD];
  cat[(size_t)bt * 1024 + d]       = (__bf16)ctx;
  cat[(size_t)bt * 1024 + 512 + d] = hs[(size_t)bt * DD + d];
}

// ---- prep kernels ----
__global__ void k_f2b(const float* __restrict__ s, __bf16* __restrict__ d, int n) {
  const int i = blockIdx.x * 256 + threadIdx.x;
  if (i < n) d[i] = (__bf16)s[i];
}

__global__ void k_embed(const int* __restrict__ words, const float* __restrict__ emb,
                        __bf16* __restrict__ x) {
  const int i = blockIdx.x * 256 + threadIdx.x;     // 1024*512
  const int bt = i >> 9, d = i & 511;
  x[i] = (__bf16)emb[(size_t)words[bt] * DD + d];
}

__global__ void k_state(const float* __restrict__ pre_h, const float* __restrict__ cell,
                        const float* __restrict__ b_ih, const float* __restrict__ b_hh,
                        __bf16* __restrict__ h2, float* __restrict__ c,
                        float* __restrict__ biasc, unsigned* __restrict__ bar) {
  const int i = blockIdx.x * 256 + threadIdx.x;
  if (i < BB * DD) { h2[i] = (__bf16)pre_h[i]; c[i] = cell[i]; }
  if (i < 2048) biasc[i] = b_ih[i] + b_hh[i];
  if (i == 0) *bar = 0u;
}

// ---------------------------------------------------------------------------
extern "C" void kernel_launch(void* const* d_in, const int* in_sizes, int n_in,
                              void* d_out, int out_size, void* d_ws, size_t ws_size,
                              hipStream_t stream)
{
  const int*   words = (const int*)  d_in[0];
  const float* enc   = (const float*)d_in[1];
  const float* pre_h = (const float*)d_in[2];
  const float* cell  = (const float*)d_in[3];
  const float* emb   = (const float*)d_in[4];
  const float* W_ih  = (const float*)d_in[5];
  const float* W_hh  = (const float*)d_in[6];
  const float* b_ih  = (const float*)d_in[7];
  const float* b_hh  = (const float*)d_in[8];
  const float* Wh_w  = (const float*)d_in[9];
  const float* Ws_w  = (const float*)d_in[10];
  const float* Ws_b  = (const float*)d_in[11];
  const float* vt_w  = (const float*)d_in[12];
  const float* V_w   = (const float*)d_in[13];
  const float* V_b   = (const float*)d_in[14];
  const float* Vp_w  = (const float*)d_in[15];
  const float* Vp_b  = (const float*)d_in[16];

  char* base = (char*)d_ws;
  size_t off = 0;
  auto alloc = [&](size_t bytes) -> void* {
    void* r = base + off;
    off = (off + bytes + 255) & ~(size_t)255;
    return r;
  };

  __bf16* x_bf    = (__bf16*)alloc((size_t)1024 * 512 * 2);
  __bf16* Wih_bf  = (__bf16*)alloc((size_t)2048 * 512 * 2);
  __bf16* Whh_bf  = (__bf16*)alloc((size_t)2048 * 512 * 2);
  __bf16* Wh_bf   = (__bf16*)alloc((size_t)512 * 512 * 2);
  __bf16* Wsw_bf  = (__bf16*)alloc((size_t)512 * 512 * 2);
  __bf16* Vw_bf   = (__bf16*)alloc((size_t)512 * 1024 * 2);
  __bf16* Vp_bf   = (__bf16*)alloc((size_t)VV * 512 * 2);
  __bf16* enc_bf  = (__bf16*)alloc((size_t)BB * SS * DD * 2);
  float*  biasc   = (float*) alloc((size_t)2048 * 4);
  float*  gates_x = (float*) alloc((size_t)1024 * 2048 * 4);
  __bf16* h2      = (__bf16*)alloc((size_t)2 * BB * DD * 2);
  float*  cstate  = (float*) alloc((size_t)BB * DD * 4);
  __bf16* hs_bf   = (__bf16*)alloc((size_t)1024 * 512 * 2);
  float*  wh      = (float*) alloc((size_t)BB * SS * DD * 4);
  float*  wsa     = (float*) alloc((size_t)1024 * 512 * 4);
  float*  attn    = (float*) alloc((size_t)1024 * SS * 4);
  __bf16* cat_bf  = (__bf16*)alloc((size_t)1024 * 1024 * 2);
  __bf16* proj_bf = (__bf16*)alloc((size_t)1024 * 512 * 2);
  unsigned* bar   = (unsigned*)alloc(256);

  // weight / input conversions to bf16
  k_f2b<<<(2048 * 512 + 255) / 256, 256, 0, stream>>>(W_ih, Wih_bf, 2048 * 512);
  k_f2b<<<(2048 * 512 + 255) / 256, 256, 0, stream>>>(W_hh, Whh_bf, 2048 * 512);
  k_f2b<<<(512 * 512 + 255) / 256, 256, 0, stream>>>(Wh_w, Wh_bf, 512 * 512);
  k_f2b<<<(512 * 512 + 255) / 256, 256, 0, stream>>>(Ws_w, Wsw_bf, 512 * 512);
  k_f2b<<<(512 * 1024 + 255) / 256, 256, 0, stream>>>(V_w, Vw_bf, 512 * 1024);
  k_f2b<<<(VV * 512 + 255) / 256, 256, 0, stream>>>(Vp_w, Vp_bf, VV * 512);
  k_f2b<<<(BB * SS * DD + 255) / 256, 256, 0, stream>>>(enc, enc_bf, BB * SS * DD);
  k_embed<<<2048, 256, 0, stream>>>(words, emb, x_bf);
  k_state<<<32, 256, 0, stream>>>(pre_h, cell, b_ih, b_hh, h2, cstate, biasc, bar);

  // gates_x[bt, 2048] = x @ W_ih^T + (b_ih + b_hh)
  k_gemm_bf16<<<dim3(2048 / 128, 1024 / 128), 256, 0, stream>>>(
      x_bf, Wih_bf, biasc, gates_x, nullptr, 1024, 2048, 512);
  // wh[b*S+s, 512] = enc @ Wh_w^T
  k_gemm_bf16<<<dim3(512 / 128, 2048 / 128), 256, 0, stream>>>(
      enc_bf, Wh_bf, nullptr, wh, nullptr, 2048, 512, 512);

  // persistent LSTM recurrence
  k_lstm<<<NWG, 256, 0, stream>>>(gates_x, Whh_bf, h2, cstate, hs_bf, bar);

  // ws[bt, 512] = hs @ Ws_w^T + Ws_b
  k_gemm_bf16<<<dim3(512 / 128, 1024 / 128), 256, 0, stream>>>(
      hs_bf, Wsw_bf, Ws_b, wsa, nullptr, 1024, 512, 512);

  // attention scores + softmax; context + concat
  k_attn<<<1024, 256, 0, stream>>>(wh, wsa, vt_w, attn);
  k_ctx<<<2048, 256, 0, stream>>>(attn, enc, hs_bf, cat_bf);

  // proj_bf[bt, 512] = cat @ V_w^T + V_b   (bf16 out only)
  k_gemm_bf16<<<dim3(512 / 128, 1024 / 128), 256, 0, stream>>>(
      cat_bf, Vw_bf, V_b, nullptr, proj_bf, 1024, 512, 1024);

  // d_out[bt, 32000] = proj @ Vp_w^T + Vp_b
  k_gemm_bf16<<<dim3(VV / 128, 1024 / 128), 256, 0, stream>>>(
      proj_bf, Vp_bf, Vp_b, (float*)d_out, nullptr, 1024, VV, 512);
}
